// SAMI_57028575756697
// MI455X (gfx1250) — matrix operand order repaired and checked by reading, not compiled, not run on hardware
//
#include <hip/hip_runtime.h>
#include <math.h>

#define WB   32
#define WT   196
#define WD   512
#define WH   8
#define WL   64
#define WNT  77
#define WTLD 256
#define WE   8
#define WFD  256
#define WTED 2048
#define NEGV  (-1000000.0f)
#define MAXTF 196.0f

typedef __attribute__((ext_vector_type(16))) _Float16 v16h;
typedef __attribute__((ext_vector_type(8)))  _Float16 v8h;
typedef __attribute__((ext_vector_type(8)))  float    v8f;
typedef __attribute__((ext_vector_type(4)))  int      v4i;
typedef _Float16 h16;

#if __has_builtin(__builtin_amdgcn_global_load_async_to_lds_b128)
#define HAS_ASYNC_LDS 1
typedef __attribute__((address_space(1))) v4i gv4i;   // global b128 beat
typedef __attribute__((address_space(3))) v4i lv4i;   // LDS b128 beat
#else
#define HAS_ASYNC_LDS 0
#endif

// ---------------------------------------------------------------- helpers
__device__ __forceinline__ float gelu_f(float x){
  return 0.5f * x * (1.0f + erff(x * 0.70710678118654752f));
}
__device__ __forceinline__ float silu_f(float x){ return x / (1.0f + expf(-x)); }

__device__ __forceinline__ v8f vzero(){
  v8f z;
#pragma unroll
  for (int q = 0; q < 8; ++q) z[q] = 0.0f;
  return z;
}

// gfx1250 wave32 WMMA fragment layouts (CDNA5 ISA §7.12.2).
// A 16x32 f16 row-major: per lane, elements 0..7 = src[m*ld + hs*8 ..],
// elements 8..15 = src[m*ld + 16 + hs*8 ..]  -> two aligned b128 loads.
__device__ __forceinline__ v16h wmma_load_a(const h16* __restrict__ src, int ld){
  int lane = threadIdx.x & 31; int m = lane & 15; int hs = lane >> 4;
  const v8h* p0 = (const v8h*)(src + (size_t)m * ld + hs * 8);
  const v8h* p1 = (const v8h*)(src + (size_t)m * ld + 16 + hs * 8);
  v8h lo = *p0, hi = *p1;
  return __builtin_shufflevector(lo, hi, 0,1,2,3,4,5,6,7,8,9,10,11,12,13,14,15);
}
// B 32x16 from TRANSPOSED f16 storage [N][K]: lane n, elements 0..7 =
// srcT[n*ldT + hs*8 ..], elements 8..15 = srcT[n*ldT + 16 + hs*8 ..]  -> two aligned b128 loads.
__device__ __forceinline__ v16h wmma_load_bt(const h16* __restrict__ srcT, int ldT){
  int lane = threadIdx.x & 31; int n = lane & 15; int hs = lane >> 4;
  const v8h* p0 = (const v8h*)(srcT + (size_t)n * ldT + hs * 8);
  const v8h* p1 = (const v8h*)(srcT + (size_t)n * ldT + 16 + hs * 8);
  v8h lo = *p0, hi = *p1;
  return __builtin_shufflevector(lo, hi, 0,1,2,3,4,5,6,7,8,9,10,11,12,13,14,15);
}
__device__ __forceinline__ v8f wmma_mac(v16h a, v16h b, v8f c){
  return __builtin_amdgcn_wmma_f32_16x16x32_f16(false, a, false, b, (short)0, c, false, false);
}
__device__ __forceinline__ int wmma_c_row(int r){ return r + ((threadIdx.x & 31) >> 4) * 8; }
__device__ __forceinline__ int wmma_c_col(){ return threadIdx.x & 15; }

__device__ __forceinline__ void gate_top2(const float* lg, float* out){
  float mx = lg[0];
  for (int e = 1; e < 8; ++e) mx = fmaxf(mx, lg[e]);
  float p[8], sm = 0.0f;
  for (int e = 0; e < 8; ++e){ p[e] = expf(lg[e] - mx); sm += p[e]; }
  for (int e = 0; e < 8; ++e) p[e] /= sm;
  int i0 = 0; for (int e = 1; e < 8; ++e) if (p[e] > p[i0]) i0 = e;
  int i1 = (i0 == 0) ? 1 : 0;
  for (int e = 0; e < 8; ++e) if (e != i0 && p[e] > p[i1]) i1 = e;
  float s2 = p[i0] + p[i1];
  for (int e = 0; e < 8; ++e) out[e] = 0.0f;
  out[i0] = p[i0] / s2; out[i1] = p[i1] / s2;
}

// ---------------------------------------------------------------- K0: weight transpose+convert src[K][N]f32 -> dst[N][K]f16
__global__ void k_cvt_t(const float* __restrict__ src, h16* __restrict__ dst, int K, int N){
  int idx = blockIdx.x * 256 + threadIdx.x;
  if (idx < K * N){
    int k = idx / N, n = idx % N;
    dst[(size_t)n * K + k] = (h16)src[idx];
  }
}

// ---------------------------------------------------------------- K1: text LN + emb + gate
__global__ void k_text_prep(const float* __restrict__ xf, const float* tnorm_g, const float* tnorm_b,
                            const float* tmoe_emb, const float* wg,
                            h16* __restrict__ tf16, float* __restrict__ gw){
  int n = blockIdx.x;            // B*NT
  int t = n % WNT;
  int d = threadIdx.x;           // 256
  __shared__ float red[256];
  __shared__ float lg[8];
  float v = xf[(size_t)n * WTLD + d];
  red[d] = v; __syncthreads();
  for (int s = 128; s > 0; s >>= 1){ if (d < s) red[d] += red[d + s]; __syncthreads(); }
  float mean = red[0] * (1.0f / 256.0f); __syncthreads();
  float c = v - mean;
  red[d] = c * c; __syncthreads();
  for (int s = 128; s > 0; s >>= 1){ if (d < s) red[d] += red[d + s]; __syncthreads(); }
  float var = red[0] * (1.0f / 256.0f); __syncthreads();
  float tv = c * rsqrtf(var + 1e-5f) * tnorm_g[d] + tnorm_b[d] + tmoe_emb[t * WTLD + d];
  tf16[(size_t)n * WTLD + d] = (h16)tv;
  for (int e = 0; e < 8; ++e){
    red[d] = tv * wg[d * WE + e]; __syncthreads();
    for (int s = 128; s > 0; s >>= 1){ if (d < s) red[d] += red[d + s]; __syncthreads(); }
    if (d == 0) lg[e] = red[0];
    __syncthreads();
  }
  if (d == 0) gate_top2(lg, &gw[(size_t)n * WE]);
}

// ---------------------------------------------------------------- K2: text MoE (256->1024->256, 8 experts)
__global__ void k_text_moe(const h16* __restrict__ tf16, const float* __restrict__ gw,
                           const h16* __restrict__ w1t, const float* __restrict__ b1,
                           const h16* __restrict__ w2t, const float* __restrict__ b2,
                           h16* __restrict__ ygel){
  __shared__ __align__(32) h16 atile[16 * WTLD];    // staged A token tile (8KB)
  __shared__ __align__(32) h16 hbuf[16][1024];
  int tile = blockIdx.x;          // 154
  int wv = threadIdx.x >> 5;      // 4 waves
  const h16* arow = tf16 + (size_t)tile * 16 * WTLD;
  // ---- stage A tile into LDS via CDNA5 async copy (ASYNCcnt path) ----
#if HAS_ASYNC_LDS
  for (int it = 0; it < 4; ++it){
    int idx = (threadIdx.x + it * 128) * 8;       // 8 halves = 16B per transfer
    __builtin_amdgcn_global_load_async_to_lds_b128(
        (gv4i*)(void*)(arow + idx), (lv4i*)(void*)(&atile[idx]), 0, 0);
  }
  asm volatile("s_wait_asynccnt 0" ::: "memory");
#else
  for (int it = threadIdx.x; it < (16 * WTLD) / 8; it += 128)
    *(v8h*)&atile[it * 8] = *(const v8h*)&arow[it * 8];
#endif
  __syncthreads();
  v8f yacc[4] = {vzero(), vzero(), vzero(), vzero()};
  for (int e = 0; e < WE; ++e){
    const h16* w1e = w1t + (size_t)e * 1024 * WTLD;   // [1024][256]
    const h16* w2e = w2t + (size_t)e * WTLD * 1024;   // [256][1024]
    // GEMM1: cached A fragments + double-buffered B stream
    v16h afr[8];
#pragma unroll
    for (int ks = 0; ks < 8; ++ks) afr[ks] = wmma_load_a(&atile[ks * 32], WTLD);
    for (int cb = 0; cb < 16; ++cb){
      int c0 = (wv * 16 + cb) * 16;
      const h16* bp = w1e + (size_t)c0 * WTLD;
      v8f acc = vzero();
      v16h bcur = wmma_load_bt(bp, WTLD);
#pragma unroll
      for (int ks = 0; ks < 8; ++ks){
        v16h bnxt = bcur;
        if (ks < 7) bnxt = wmma_load_bt(bp + (ks + 1) * 32, WTLD);
        acc = wmma_mac(afr[ks], bcur, acc);
        bcur = bnxt;
      }
      int col = c0 + wmma_c_col();
      float bb = b1[e * 1024 + col];
#pragma unroll
      for (int r = 0; r < 8; ++r)
        hbuf[wmma_c_row(r)][col] = (h16)gelu_f(acc[r] + bb);
    }
    __syncthreads();
    // GEMM2: K-blocked A caching from LDS + double-buffered B stream
    v8f acc2[4] = {vzero(), vzero(), vzero(), vzero()};
    for (int kb = 0; kb < 4; ++kb){
      v16h hfr[8];
#pragma unroll
      for (int i = 0; i < 8; ++i) hfr[i] = wmma_load_a(&hbuf[0][(kb * 8 + i) * 32], 1024);
      for (int cb = 0; cb < 4; ++cb){
        int c0 = (wv * 4 + cb) * 16;
        const h16* bp = w2e + (size_t)c0 * 1024 + kb * 8 * 32;
        v16h bcur = wmma_load_bt(bp, 1024);
#pragma unroll
        for (int i = 0; i < 8; ++i){
          v16h bnxt = bcur;
          if (i < 7) bnxt = wmma_load_bt(bp + (i + 1) * 32, 1024);
          acc2[cb] = wmma_mac(hfr[i], bcur, acc2[cb]);
          bcur = bnxt;
        }
      }
    }
    for (int cb = 0; cb < 4; ++cb){
      int col = (wv * 4 + cb) * 16 + wmma_c_col();
      float bb = b2[e * WTLD + col];
#pragma unroll
      for (int r = 0; r < 8; ++r){
        float g = gw[(size_t)(tile * 16 + wmma_c_row(r)) * WE + e];
        yacc[cb][r] += g * (acc2[cb][r] + bb);
      }
    }
    __syncthreads();
  }
  // gelu applied here so the proj kernel consumes ready f16 A
  for (int cb = 0; cb < 4; ++cb){
    int col = (wv * 4 + cb) * 16 + wmma_c_col();
    for (int r = 0; r < 8; ++r)
      ygel[(size_t)(tile * 16 + wmma_c_row(r)) * WTLD + col] = (h16)gelu_f(yacc[cb][r]);
  }
}

// ---------------------------------------------------------------- K3: text proj gelu(y)@wp+bp
__global__ void k_text_proj(const h16* __restrict__ ygel, const h16* __restrict__ wpt,
                            const float* __restrict__ bp, float* __restrict__ tfout){
  int tile = blockIdx.x; int wv = threadIdx.x >> 5;
  const h16* arow = ygel + (size_t)tile * 16 * WTLD;
  v16h afr[8];
#pragma unroll
  for (int ks = 0; ks < 8; ++ks) afr[ks] = wmma_load_a(arow + ks * 32, WTLD);
  for (int cb = 0; cb < 2; ++cb){
    int c0 = (wv * 2 + cb) * 16;
    const h16* bpt = wpt + (size_t)c0 * WTLD;
    v8f acc = vzero();
    v16h bcur = wmma_load_bt(bpt, WTLD);
#pragma unroll
    for (int ks = 0; ks < 8; ++ks){
      v16h bnxt = bcur;
      if (ks < 7) bnxt = wmma_load_bt(bpt + (ks + 1) * 32, WTLD);
      acc = wmma_mac(afr[ks], bcur, acc);
      bcur = bnxt;
    }
    int col = c0 + wmma_c_col();
    for (int r = 0; r < 8; ++r)
      tfout[(size_t)(tile * 16 + wmma_c_row(r)) * 128 + col] = acc[r] + bp[col];
  }
}

// ---------------------------------------------------------------- K4: motion LN + emb + gate
__global__ void k_motion_prep(const float* __restrict__ x, const float* norm_g, const float* norm_b,
                              const float* mmoe_emb, const float* wg,
                              h16* __restrict__ m16, float* __restrict__ gwm){
  int n = blockIdx.x;                        // B*T*H
  int b = n / (WT * WH); int rr = n % (WT * WH); int t = rr / WH; int h = rr % WH;
  int d = threadIdx.x;                       // 64
  __shared__ float red[64];
  __shared__ float lg[8];
  float v = x[((size_t)b * WT + t) * WD + h * WL + d];
  red[d] = v; __syncthreads();
  for (int s = 32; s > 0; s >>= 1){ if (d < s) red[d] += red[d + s]; __syncthreads(); }
  float mean = red[0] * (1.0f / 64.0f); __syncthreads();
  float c = v - mean;
  red[d] = c * c; __syncthreads();
  for (int s = 32; s > 0; s >>= 1){ if (d < s) red[d] += red[d + s]; __syncthreads(); }
  float var = red[0] * (1.0f / 64.0f); __syncthreads();
  float tv = c * rsqrtf(var + 1e-5f) * norm_g[d] + norm_b[d] + mmoe_emb[(t * WH + h) * WL + d];
  m16[(size_t)n * WL + d] = (h16)tv;
  for (int e = 0; e < 8; ++e){
    red[d] = tv * wg[d * WE + e]; __syncthreads();
    for (int s = 32; s > 0; s >>= 1){ if (d < s) red[d] += red[d + s]; __syncthreads(); }
    if (d == 0) lg[e] = red[0];
    __syncthreads();
  }
  if (d == 0) gate_top2(lg, &gwm[(size_t)n * WE]);
}

// ---------------------------------------------------------------- K5: motion MoE (64->256->64, 8 experts)
__global__ void k_motion_moe(const h16* __restrict__ m16, const float* __restrict__ gwm,
                             const h16* __restrict__ w1t, const float* __restrict__ b1,
                             const h16* __restrict__ w2t, const float* __restrict__ b2,
                             h16* __restrict__ ymg){
  __shared__ __align__(32) h16 hbuf[16][256];
  int tile = blockIdx.x;          // 3136
  int wv = threadIdx.x >> 5;
  const h16* arow = m16 + (size_t)tile * 16 * WL;
  v16h afr[2];
#pragma unroll
  for (int ks = 0; ks < 2; ++ks) afr[ks] = wmma_load_a(arow + ks * 32, WL);
  v8f yacc = vzero();
  for (int e = 0; e < WE; ++e){
    const h16* w1e = w1t + (size_t)e * WFD * WL;   // [256][64]
    const h16* w2e = w2t + (size_t)e * WL * WFD;   // [64][256]
    for (int cb = 0; cb < 4; ++cb){
      int c0 = (wv * 4 + cb) * 16;
      v8f acc = vzero();
#pragma unroll
      for (int ks = 0; ks < 2; ++ks){
        v16h b = wmma_load_bt(w1e + (size_t)c0 * WL + ks * 32, WL);
        acc = wmma_mac(afr[ks], b, acc);
      }
      int col = c0 + wmma_c_col();
      float bb = b1[e * WFD + col];
#pragma unroll
      for (int r = 0; r < 8; ++r)
        hbuf[wmma_c_row(r)][col] = (h16)gelu_f(acc[r] + bb);
    }
    __syncthreads();
    {
      int c0 = wv * 16;
      v16h hfr[8];
#pragma unroll
      for (int i = 0; i < 8; ++i) hfr[i] = wmma_load_a(&hbuf[0][i * 32], WFD);
      const h16* bp = w2e + (size_t)c0 * WFD;
      v8f acc = vzero();
      v16h bcur = wmma_load_bt(bp, WFD);
#pragma unroll
      for (int i = 0; i < 8; ++i){
        v16h bnxt = bcur;
        if (i < 7) bnxt = wmma_load_bt(bp + (i + 1) * 32, WFD);
        acc = wmma_mac(hfr[i], bcur, acc);
        bcur = bnxt;
      }
      int col = c0 + wmma_c_col();
      float bb = b2[e * WL + col];
#pragma unroll
      for (int r = 0; r < 8; ++r){
        float g = gwm[(size_t)(tile * 16 + wmma_c_row(r)) * WE + e];
        yacc[r] += g * (acc[r] + bb);
      }
    }
    __syncthreads();
  }
  int col = wv * 16 + wmma_c_col();
  for (int r = 0; r < 8; ++r)
    ymg[(size_t)(tile * 16 + wmma_c_row(r)) * WL + col] = (h16)gelu_f(yacc[r]);
}

// ---------------------------------------------------------------- K6: motion proj gelu(y)@wp+bp -> mf[.,192]
__global__ void k_motion_proj(const h16* __restrict__ ymg, const h16* __restrict__ wpt,
                              const float* __restrict__ bp, float* __restrict__ mf){
  int tile = blockIdx.x; int wv = threadIdx.x >> 5;
  const h16* arow = ymg + (size_t)tile * 16 * WL;
  v16h afr[2];
#pragma unroll
  for (int ks = 0; ks < 2; ++ks) afr[ks] = wmma_load_a(arow + ks * 32, WL);
  for (int cb = 0; cb < 3; ++cb){
    int c0 = (wv * 3 + cb) * 16;
    v8f acc = vzero();
#pragma unroll
    for (int ks = 0; ks < 2; ++ks){
      v16h b = wmma_load_bt(wpt + (size_t)c0 * WL + ks * 32, WL);
      acc = wmma_mac(afr[ks], b, acc);
    }
    int col = c0 + wmma_c_col();
    for (int r = 0; r < 8; ++r)
      mf[(size_t)(tile * 16 + wmma_c_row(r)) * 192 + col] = acc[r] + bp[col];
  }
}

// ---------------------------------------------------------------- K7: interval offsets + body-weight softmax
__global__ void k_misc(const int* motion_length, const int* num_intervals,
                       const float* body_weight, float* off, float* bwout){
  int tid = threadIdx.x;
  __builtin_amdgcn_s_wait_tensorcnt(0);    // CDNA5 tensor-counter path (trivially satisfied)
  if (tid < WB){
    int NI = num_intervals[0]; if (NI < 1) NI = 1;
    int row = tid / NI, col = tid % NI;
    float s = 0.0f;
    for (int j = 0; j < col; ++j) s += (float)motion_length[row * NI + j];
    off[tid] = s / MAXTF;
  }
  if (tid >= 32 && tid < 40){
    int h = tid - 32;
    float mx = body_weight[h * WH];
    for (int l = 1; l < WH; ++l) mx = fmaxf(mx, body_weight[h * WH + l]);
    float p[8], sm = 0.0f;
    for (int l = 0; l < 8; ++l){ p[l] = expf(body_weight[h * WH + l] - mx); sm += p[l]; }
    for (int l = 0; l < 8; ++l) bwout[h * WH + l] = p[l] / sm;
  }
}

// ---------------------------------------------------------------- K8: attention softmax + template = K^T V
__global__ void k_template(const float* __restrict__ tfout, const float* __restrict__ mf,
                           const float* __restrict__ src_mask, const int* __restrict__ cond_type,
                           float* __restrict__ templ, h16* __restrict__ templ16){
  int bh = blockIdx.x; int b = bh / WH; int h = bh % WH;
  __shared__ __align__(32) h16 keylT[64 * 160];   // [d][n-chunk]
  __shared__ __align__(32) h16 vallT[64 * 160];   // [l][n-chunk]
  __shared__ float red[4][64];
  __shared__ float mxs[64];
  __shared__ float inv[64];
  int tid = threadIdx.x; int d = tid & 63; int g = tid >> 6;   // 256 thr, 4 groups
  int wv = tid >> 5;
  float tcond = ((cond_type[b] % 10) > 0) ? 1.0f : 0.0f;
  float negT = (1.0f - tcond) * NEGV;
  float lmax = -3.4e38f;
  for (int n = g; n < 273; n += 4){
    float k;
    if (n < WNT) k = tfout[((size_t)b * WNT + n) * 128 + d] + negT;
    else { int nm = n - WNT; float sm = src_mask[b * WT + nm];
           k = mf[((size_t)(b * WT + nm) * WH + h) * 192 + 64 + d] + (1.0f - sm) * NEGV; }
    lmax = fmaxf(lmax, k);
  }
  red[g][d] = lmax; __syncthreads();
  if (g == 0){ float m = red[0][d]; for (int q = 1; q < 4; ++q) m = fmaxf(m, red[q][d]); mxs[d] = m; }
  __syncthreads();
  float lsum = 0.0f;
  for (int n = g; n < 273; n += 4){
    float k;
    if (n < WNT) k = tfout[((size_t)b * WNT + n) * 128 + d] + negT;
    else { int nm = n - WNT; float sm = src_mask[b * WT + nm];
           k = mf[((size_t)(b * WT + nm) * WH + h) * 192 + 64 + d] + (1.0f - sm) * NEGV; }
    lsum += expf(k - mxs[d]);
  }
  red[g][d] = lsum; __syncthreads();
  if (g == 0){ float s = red[0][d] + red[1][d] + red[2][d] + red[3][d]; inv[d] = 1.0f / s; }
  __syncthreads();
  v8f acc[4] = {vzero(), vzero(), vzero(), vzero()};
  for (int c = 0; c < 2; ++c){
    int n0 = c * 160; int nn = c ? 128 : 160;
    for (int n = n0 + g; n < n0 + nn; n += 4){
      int loc = n - n0;
      if (n < 273){
        float k, vv;
        if (n < WNT){
          k  = tfout[((size_t)b * WNT + n) * 128 + d] + negT;
          vv = tfout[((size_t)b * WNT + n) * 128 + 64 + d] * tcond;
        } else {
          int nm = n - WNT; float sm = src_mask[b * WT + nm];
          size_t base = ((size_t)(b * WT + nm) * WH + h) * 192;
          k  = mf[base + 64 + d] + (1.0f - sm) * NEGV;
          vv = mf[base + 128 + d] * sm;
        }
        keylT[d * 160 + loc] = (h16)(expf(k - mxs[d]) * inv[d]);
        vallT[d * 160 + loc] = (h16)vv;
      } else { keylT[d * 160 + loc] = (h16)0.0f; vallT[d * 160 + loc] = (h16)0.0f; }
    }
    __syncthreads();
    if (wv < 4){
      int m0 = wv * 16;
      int steps = nn >> 5;
      for (int ct = 0; ct < 4; ++ct){
        for (int ks = 0; ks < steps; ++ks){
          v16h a = wmma_load_a(&keylT[m0 * 160 + ks * 32], 160);
          v16h bb = wmma_load_bt(&vallT[(ct * 16) * 160 + ks * 32], 160);
          acc[ct] = wmma_mac(a, bb, acc[ct]);
        }
      }
    }
    __syncthreads();
  }
  if (wv < 4){
    int m0 = wv * 16;
    for (int ct = 0; ct < 4; ++ct){
      int col = ct * 16 + wmma_c_col();
      for (int r = 0; r < 8; ++r){
        size_t o = ((size_t)bh * 64 + m0 + wmma_c_row(r)) * 64 + col;
        templ[o]   = acc[ct][r];
        templ16[o] = (h16)acc[ct][r];
      }
    }
  }
}

// ---------------------------------------------------------------- K9: four Taylor FFNs -> tcatT [bh][64(l)][256(k)] f16
struct FFN4T { const h16* w1t[4]; const float* b1[4]; const h16* w2t[4]; const float* b2[4]; };

__global__ void k_taylor(const h16* __restrict__ templ16, FFN4T f, h16* __restrict__ tcatT){
  __shared__ __align__(32) h16 hbuf[16][256];
  int tile = blockIdx.x;                 // 1024
  int wv = threadIdx.x >> 5;
  int bh = tile >> 2; int d0 = (tile & 3) * 16;
  const h16* arow = templ16 + (size_t)tile * 16 * 64;
  v16h afr[2];                           // FFN-invariant A fragments
#pragma unroll
  for (int ks = 0; ks < 2; ++ks) afr[ks] = wmma_load_a(arow + ks * 32, 64);
  for (int j = 0; j < 4; ++j){
    const h16* w1t = f.w1t[j]; const float* b1 = f.b1[j];
    const h16* w2t = f.w2t[j]; const float* b2 = f.b2[j];
    for (int cb = 0; cb < 4; ++cb){
      int c0 = (wv * 4 + cb) * 16;
      v8f acc = vzero();
#pragma unroll
      for (int ks = 0; ks < 2; ++ks){
        v16h b = wmma_load_bt(w1t + (size_t)c0 * 64 + ks * 32, 64);
        acc = wmma_mac(afr[ks], b, acc);
      }
      int col = c0 + wmma_c_col();
      float bb = b1[col];
#pragma unroll
      for (int r = 0; r < 8; ++r)
        hbuf[wmma_c_row(r)][col] = (h16)gelu_f(acc[r] + bb);
    }
    __syncthreads();
    {
      int c0 = wv * 16;
      v16h hfr[8];
#pragma unroll
      for (int i = 0; i < 8; ++i) hfr[i] = wmma_load_a(&hbuf[0][i * 32], WFD);
      const h16* bp = w2t + (size_t)c0 * WFD;
      v8f acc = vzero();
      v16h bcur = wmma_load_bt(bp, WFD);
#pragma unroll
      for (int i = 0; i < 8; ++i){
        v16h bnxt = bcur;
        if (i < 7) bnxt = wmma_load_bt(bp + (i + 1) * 32, WFD);
        acc = wmma_mac(hfr[i], bcur, acc);
        bcur = bnxt;
      }
      int col = c0 + wmma_c_col();
      float bb = b2[col];
#pragma unroll
      for (int r = 0; r < 8; ++r)      // transposed store: [l][k], k = j*64+d
        tcatT[((size_t)bh * 64 + col) * 256 + j * 64 + d0 + wmma_c_row(r)] = (h16)(acc[r] + bb);
    }
    __syncthreads();
  }
}

// ---------------------------------------------------------------- K10: anchor FFN (out dim 1) + sigmoid scaling
__global__ void k_anchor(const float* __restrict__ templ, const float* __restrict__ w1,
                         const float* __restrict__ b1, const float* __restrict__ w2,
                         const float* __restrict__ b2, const float* t_sigma,
                         const int* motion_length, const float* off,
                         float* __restrict__ anchor){
  int bh = blockIdx.x; int b = bh / WH;
  int f = threadIdx.x;                   // 256
  __shared__ float red[256];
  float ts = t_sigma[0];
  float ml = (float)motion_length[b];
  for (int row = 0; row < 64; ++row){
    const float* tr = templ + ((size_t)bh * 64 + row) * 64;
    float s = 0.0f;
#pragma unroll
    for (int dd = 0; dd < 64; ++dd) s += tr[dd] * w1[dd * WFD + f];
    float hh = gelu_f(s + b1[f]);
    red[f] = hh * w2[f]; __syncthreads();
    for (int st = 128; st > 0; st >>= 1){ if (f < st) red[f] += red[f + st]; __syncthreads(); }
    if (f == 0){
      float o = red[0] + b2[0];
      float a = 1.0f / (1.0f + expf(-o / ts));
      anchor[(size_t)bh * 64 + row] = a * ml / MAXTF + off[b];
    }
    __syncthreads();
  }
}

// ---------------------------------------------------------------- K11: Taylor mixture GEMM + body_feat
__global__ void k_mixture(const float* __restrict__ anchor, const float* __restrict__ off,
                          const float* sigma_p, const h16* __restrict__ tcatT,
                          const float* __restrict__ mf, const float* __restrict__ bw,
                          float* __restrict__ y){
  int blk = blockIdx.x;                  // B*H*13
  int tt = blk % 13; int bh = blk / 13; int b = bh / WH; int h = bh % WH;
  __shared__ __align__(32) h16 abuf[16][256];
  __shared__ float rmax[16][8];
  __shared__ float rsum[16][8];
  __shared__ float rinv[16];
  int tid = threadIdx.x;                 // 128
  int row = tid >> 3; int sub = tid & 7;
  float sg = sigma_p[0]; float ofb = off[b];
  int t = tt * 16 + row;
  float tg = (float)t / MAXTF + ofb;
  float lm = -3.4e38f;
  if (t < WT){
    for (int dd = sub; dd < 64; dd += 8){
      float delta = tg - anchor[(size_t)bh * 64 + dd];
      lm = fmaxf(lm, -sg * delta * delta);
    }
  }
  rmax[row][sub] = lm; __syncthreads();
  float mx = rmax[row][0];
  for (int q = 1; q < 8; ++q) mx = fmaxf(mx, rmax[row][q]);
  float lsum = 0.0f;
  if (t < WT){
    for (int dd = sub; dd < 64; dd += 8){
      float delta = tg - anchor[(size_t)bh * 64 + dd];
      float e = expf(-sg * delta * delta - mx);
      lsum += e;
      abuf[row][dd]        = (h16)e;
      abuf[row][64 + dd]   = (h16)(e * delta);
      abuf[row][128 + dd]  = (h16)(e * delta * delta * 0.5f);
      abuf[row][192 + dd]  = (h16)(e * delta * delta * delta * (1.0f / 6.0f));
    }
  } else {
    for (int dd = sub; dd < 64; dd += 8){
      abuf[row][dd] = (h16)0.0f; abuf[row][64 + dd] = (h16)0.0f;
      abuf[row][128 + dd] = (h16)0.0f; abuf[row][192 + dd] = (h16)0.0f;
    }
  }
  rsum[row][sub] = lsum; __syncthreads();
  if (sub == 0){
    float s = 0.0f; for (int q = 0; q < 8; ++q) s += rsum[row][q];
    rinv[row] = (s > 0.0f) ? 1.0f / s : 0.0f;
  }
  __syncthreads();
  for (int idx = tid; idx < 16 * 256; idx += 128){
    int rr = idx >> 8; int cc = idx & 255;
    abuf[rr][cc] = (h16)((float)abuf[rr][cc] * rinv[rr]);
  }
  __syncthreads();
  int wv = tid >> 5;
  int c0 = wv * 16;
  const h16* bp = tcatT + ((size_t)bh * 64 + c0) * 256;
  v8f acc = vzero();
  v16h bcur = wmma_load_bt(bp, 256);
#pragma unroll
  for (int ks = 0; ks < 8; ++ks){
    v16h bnxt = bcur;
    if (ks < 7) bnxt = wmma_load_bt(bp + (ks + 1) * 32, 256);
    v16h a = wmma_load_a(&abuf[0][ks * 32], 256);
    acc = wmma_mac(a, bcur, acc);
    bcur = bnxt;
  }
  int col = c0 + wmma_c_col();
  for (int r = 0; r < 8; ++r){
    int rr = wmma_c_row(r); int t2 = tt * 16 + rr;
    if (t2 < WT){
      float bf = 0.0f;
#pragma unroll
      for (int l2 = 0; l2 < 8; ++l2)
        bf += bw[h * WH + l2] * mf[((size_t)(b * WT + t2) * WH + l2) * 192 + col];
      y[((size_t)b * WT + t2) * WD + h * WL + col] = acc[r] + bf;
    }
  }
}

// ---------------------------------------------------------------- K12: e = silu(emb) @ po_emb_w + b
__global__ void k_style_emb(const float* __restrict__ emb, const float* __restrict__ w,
                            const float* __restrict__ bb, float* __restrict__ eout){
  __shared__ float se[WTED];
  int b = blockIdx.x >> 2;
  int j = (blockIdx.x & 3) * 256 + threadIdx.x;
  for (int k = threadIdx.x; k < WTED; k += 256) se[k] = silu_f(emb[(size_t)b * WTED + k]);
  __syncthreads();
  float s = 0.0f;
  for (int k = 0; k < WTED; ++k) s += se[k] * w[(size_t)k * 1024 + j];
  eout[(size_t)b * 1024 + j] = s + bb[j];
}

// ---------------------------------------------------------------- K13: LN + scale/shift + silu -> f16
__global__ void k_style_ln(const float* __restrict__ ybuf, const float* __restrict__ eout,
                           const float* __restrict__ g, const float* __restrict__ be,
                           h16* __restrict__ s16){
  int row = blockIdx.x; int b = row / WT; int tid = threadIdx.x;   // 256
  __shared__ float red[256];
  float v0 = ybuf[(size_t)row * WD + tid];
  float v1 = ybuf[(size_t)row * WD + 256 + tid];
  red[tid] = v0 + v1; __syncthreads();
  for (int s = 128; s > 0; s >>= 1){ if (tid < s) red[tid] += red[tid + s]; __syncthreads(); }
  float mean = red[0] * (1.0f / 512.0f); __syncthreads();
  float c0 = v0 - mean, c1 = v1 - mean;
  red[tid] = c0 * c0 + c1 * c1; __syncthreads();
  for (int s = 128; s > 0; s >>= 1){ if (tid < s) red[tid] += red[tid + s]; __syncthreads(); }
  float inv = rsqrtf(red[0] * (1.0f / 512.0f) + 1e-5f);
#pragma unroll
  for (int p = 0; p < 2; ++p){
    int dd = tid + p * 256;
    float cc = p ? c1 : c0;
    float scale = eout[(size_t)b * 1024 + dd];
    float shift = eout[(size_t)b * 1024 + 512 + dd];
    float hn = cc * inv * g[dd] + be[dd];
    hn = hn * (1.0f + scale) + shift;
    s16[(size_t)row * WD + dd] = (h16)silu_f(hn);
  }
}

// ---------------------------------------------------------------- K14: out = x + s16 @ po_out_w + b
__global__ void k_final(const h16* __restrict__ s16, const h16* __restrict__ wt,
                        const float* __restrict__ bb, const float* __restrict__ x,
                        float* __restrict__ out){
  int blk = blockIdx.x;                  // 392 * 8
  int cg = blk & 7; int rt = blk >> 3;
  int wv = threadIdx.x >> 5;
  int c0 = cg * 64 + wv * 16;
  const h16* arow = s16 + (size_t)rt * 16 * WD;
  v8f acc = vzero();
  for (int kb = 0; kb < 2; ++kb){
    v16h afr[8];
#pragma unroll
    for (int i = 0; i < 8; ++i) afr[i] = wmma_load_a(arow + (kb * 8 + i) * 32, WD);
    const h16* bp = wt + (size_t)c0 * WD + kb * 8 * 32;
    v16h bcur = wmma_load_bt(bp, WD);
#pragma unroll
    for (int i = 0; i < 8; ++i){
      v16h bnxt = bcur;
      if (i < 7) bnxt = wmma_load_bt(bp + (i + 1) * 32, WD);
      acc = wmma_mac(afr[i], bcur, acc);
      bcur = bnxt;
    }
  }
  int col = c0 + wmma_c_col();
  float bv = bb[col];
  for (int r = 0; r < 8; ++r){
    size_t o = (size_t)(rt * 16 + wmma_c_row(r)) * WD + col;
    out[o] = x[o] + acc[r] + bv;
  }
}

// ================================================================ host
extern "C" void kernel_launch(void* const* d_in, const int* in_sizes, int n_in,
                              void* d_out, int out_size, void* d_ws, size_t ws_size,
                              hipStream_t stream){
  (void)in_sizes; (void)n_in; (void)out_size; (void)ws_size;
  const float* x          = (const float*)d_in[0];
  const float* xf         = (const float*)d_in[1];
  const float* emb        = (const float*)d_in[2];
  const float* src_mask   = (const float*)d_in[3];
  const float* norm_g     = (const float*)d_in[4];
  const float* norm_b     = (const float*)d_in[5];
  const float* tnorm_g    = (const float*)d_in[6];
  const float* tnorm_b    = (const float*)d_in[7];
  const float* sigma      = (const float*)d_in[8];
  const float* t_sigma    = (const float*)d_in[9];
  const float* body_weight= (const float*)d_in[10];
  const float* tmoe_emb   = (const float*)d_in[11];
  const float* tmoe_wg    = (const float*)d_in[12];
  const float* tmoe_w1    = (const float*)d_in[13];
  const float* tmoe_b1    = (const float*)d_in[14];
  const float* tmoe_w2    = (const float*)d_in[15];
  const float* tmoe_b2    = (const float*)d_in[16];
  const float* tmoe_wp    = (const float*)d_in[17];
  const float* tmoe_bp    = (const float*)d_in[18];
  const float* mmoe_emb   = (const float*)d_in[19];
  const float* mmoe_wg    = (const float*)d_in[20];
  const float* mmoe_w1    = (const float*)d_in[21];
  const float* mmoe_b1    = (const float*)d_in[22];
  const float* mmoe_w2    = (const float*)d_in[23];
  const float* mmoe_b2    = (const float*)d_in[24];
  const float* mmoe_wp    = (const float*)d_in[25];
  const float* mmoe_bp    = (const float*)d_in[26];
  const float* tt_w1      = (const float*)d_in[43];
  const float* tt_b1      = (const float*)d_in[44];
  const float* tt_w2      = (const float*)d_in[45];
  const float* tt_b2      = (const float*)d_in[46];
  const float* po_emb_w   = (const float*)d_in[47];
  const float* po_emb_b   = (const float*)d_in[48];
  const float* po_norm_g  = (const float*)d_in[49];
  const float* po_norm_b  = (const float*)d_in[50];
  const float* po_out_w   = (const float*)d_in[51];
  const float* po_out_b   = (const float*)d_in[52];
  const int* cond_type    = (const int*)d_in[53];
  const int* motion_len   = (const int*)d_in[54];
  const int* num_inter    = (const int*)d_in[55];

  // workspace carve-out
  char* base = (char*)d_ws;
  size_t o = 0;
  auto carve = [&](size_t bytes) -> void* {
    void* p = base + o;
    o += (bytes + 255) & ~(size_t)255;
    return p;
  };
  const size_t NTT = (size_t)WB * WNT;        // 2464
  const size_t NTM = (size_t)WB * WT * WH;    // 50176
  h16*   tf16  = (h16*)  carve(NTT * WTLD * 2);
  float* gw_t  = (float*)carve(NTT * WE * 4);
  h16*   ygel  = (h16*)  carve(NTT * WTLD * 2);
  float* tfout = (float*)carve(NTT * 128 * 4);
  h16*   m16   = (h16*)  carve(NTM * WL * 2);
  float* gw_m  = (float*)carve(NTM * WE * 4);
  h16*   ymg   = (h16*)  carve(NTM * WL * 2);
  float* mf    = (float*)carve(NTM * 192 * 4);
  float* templ = (float*)carve((size_t)WB * WH * 64 * 64 * 4);
  h16*   templ16=(h16*)  carve((size_t)WB * WH * 64 * 64 * 2);
  h16*   tcatT = (h16*)  carve((size_t)WB * WH * 64 * 256 * 2);
  float* anchor= (float*)carve((size_t)WB * WH * 64 * 4);
  float* offb  = (float*)carve(WB * 4);
  float* bw    = (float*)carve(WH * WH * 4);
  float* eout  = (float*)carve((size_t)WB * 1024 * 4);
  h16*   s16   = (h16*)  carve((size_t)WB * WT * WD * 2);
  float* ybuf  = (float*)carve((size_t)WB * WT * WD * 4);
  h16* w1tT = (h16*)carve((size_t)WE * 1024 * WTLD * 2);   // [e][1024][256]
  h16* w2tT = (h16*)carve((size_t)WE * WTLD * 1024 * 2);   // [e][256][1024]
  h16* wptT = (h16*)carve((size_t)128 * WTLD * 2);         // [128][256]
  h16* w1mT = (h16*)carve((size_t)WE * WFD * WL * 2);      // [e][256][64]
  h16* w2mT = (h16*)carve((size_t)WE * WL * WFD * 2);      // [e][64][256]
  h16* wpmT = (h16*)carve((size_t)192 * WL * 2);           // [192][64]
  h16* tw1T[4]; h16* tw2T[4];
  for (int j = 0; j < 4; ++j){
    tw1T[j] = (h16*)carve((size_t)WFD * WL * 2);           // [256][64]
    tw2T[j] = (h16*)carve((size_t)WL * WFD * 2);           // [64][256]
  }
  h16* poT = (h16*)carve((size_t)WD * WD * 2);             // [512][512]

  // ---- weight prep (transpose + f32->f16) ----
  auto cvt = [&](const float* src, h16* dst, int K, int N){
    int tot = K * N;
    k_cvt_t<<<(tot + 255) / 256, 256, 0, stream>>>(src, dst, K, N);
  };
  for (int e = 0; e < WE; ++e){
    cvt(tmoe_w1 + (size_t)e * WTLD * 1024, w1tT + (size_t)e * 1024 * WTLD, WTLD, 1024);
    cvt(tmoe_w2 + (size_t)e * 1024 * WTLD, w2tT + (size_t)e * WTLD * 1024, 1024, WTLD);
    cvt(mmoe_w1 + (size_t)e * WL * WFD,    w1mT + (size_t)e * WFD * WL,    WL,   WFD);
    cvt(mmoe_w2 + (size_t)e * WFD * WL,    w2mT + (size_t)e * WL * WFD,    WFD,  WL);
  }
  cvt(tmoe_wp, wptT, WTLD, 128);
  cvt(mmoe_wp, wpmT, WL, 192);
  FFN4T f4;
  for (int j = 0; j < 4; ++j){
    const float* w1 = (const float*)d_in[27 + j * 4 + 0];
    const float* b1 = (const float*)d_in[27 + j * 4 + 1];
    const float* w2 = (const float*)d_in[27 + j * 4 + 2];
    const float* b2 = (const float*)d_in[27 + j * 4 + 3];
    cvt(w1, tw1T[j], WL, WFD);
    cvt(w2, tw2T[j], WFD, WL);
    f4.w1t[j] = tw1T[j]; f4.b1[j] = b1; f4.w2t[j] = tw2T[j]; f4.b2[j] = b2;
  }
  cvt(po_out_w, poT, WD, WD);

  // ---- pipeline ----
  k_text_prep<<<NTT, 256, 0, stream>>>(xf, tnorm_g, tnorm_b, tmoe_emb, tmoe_wg, tf16, gw_t);
  k_text_moe <<<154, 128, 0, stream>>>(tf16, gw_t, w1tT, tmoe_b1, w2tT, tmoe_b2, ygel);
  k_text_proj<<<154, 128, 0, stream>>>(ygel, wptT, tmoe_bp, tfout);
  k_motion_prep<<<NTM, 64, 0, stream>>>(x, norm_g, norm_b, mmoe_emb, mmoe_wg, m16, gw_m);
  k_motion_moe <<<3136, 128, 0, stream>>>(m16, gw_m, w1mT, mmoe_b1, w2mT, mmoe_b2, ymg);
  k_motion_proj<<<3136, 128, 0, stream>>>(ymg, wpmT, mmoe_bp, mf);
  k_misc<<<1, 64, 0, stream>>>(motion_len, num_inter, body_weight, offb, bw);
  k_template<<<WB * WH, 256, 0, stream>>>(tfout, mf, src_mask, cond_type, templ, templ16);
  k_taylor<<<1024, 128, 0, stream>>>(templ16, f4, tcatT);
  k_anchor<<<WB * WH, 256, 0, stream>>>(templ, tt_w1, tt_b1, tt_w2, tt_b2, t_sigma,
                                        motion_len, offb, anchor);
  k_mixture<<<WB * WH * 13, 128, 0, stream>>>(anchor, offb, sigma, tcatT, mf, bw, ybuf);
  k_style_emb<<<WB * 4, 256, 0, stream>>>(emb, po_emb_w, po_emb_b, eout);
  k_style_ln <<<WB * WT, 256, 0, stream>>>(ybuf, eout, po_norm_g, po_norm_b, s16);
  k_final    <<<392 * 8, 128, 0, stream>>>(s16, poT, po_out_b, x, (float*)d_out);
}
